// DynamicRouter_47296179863781
// MI455X (gfx1250) — matrix operand order repaired and checked by reading, not compiled, not run on hardware
//
#include <hip/hip_runtime.h>

#define GEMM_ASYNC 1   // set to 0 if global_load_async_to_lds asm fails to assemble

typedef __attribute__((ext_vector_type(16))) __bf16 v16bf;
typedef __attribute__((ext_vector_type(8)))  float  v8f;

union BF16Frag {
    v16bf v;
    uint4 q[2];
    unsigned short u[16];
};

__device__ inline v8f v8f_zero() {
    v8f z;
#pragma unroll
    for (int i = 0; i < 8; i++) z[i] = 0.f;
    return z;
}

__device__ inline v8f wmma_bf16(const BF16Frag& a, const BF16Frag& b, v8f c) {
    return __builtin_amdgcn_wmma_f32_16x16x32_bf16(false, a.v, false, b.v,
                                                   (short)0, c, false, false);
}

// float -> bf16 (round to nearest even; inputs finite)
__device__ inline unsigned short f2bf(float f) {
    unsigned u = __float_as_uint(f);
    u += 0x7FFFu + ((u >> 16) & 1u);
    return (unsigned short)(u >> 16);
}

// order-preserving float<->uint key (for atomicMax-based max pooling)
__device__ inline unsigned fkey(float f) {
    unsigned u = __float_as_uint(f);
    return (u & 0x80000000u) ? ~u : (u | 0x80000000u);
}
__device__ inline float dekey(unsigned k) {
    unsigned u = (k & 0x80000000u) ? (k & 0x7FFFFFFFu) : ~k;
    return __uint_as_float(u);
}

// --- CDNA5 async global->LDS copy (ASYNCcnt-tracked) ---------------------
typedef __attribute__((address_space(3))) const void lds_cv_t;

__device__ inline unsigned lds_off(const void* p) {
    return (unsigned)(unsigned long long)(lds_cv_t*)p;  // AS3 ptr == LDS byte offset
}

__device__ inline void async_ld_b128(unsigned dst_lds, const void* src_global) {
    asm volatile("global_load_async_to_lds_b128 %0, %1, off"
                 :: "v"(dst_lds), "v"(src_global)
                 : "memory");
}
__device__ inline void wait_async_le(int) = delete;
__device__ inline void wait_async_6() { asm volatile("s_wait_asynccnt 0x6" ::: "memory"); }
__device__ inline void wait_async_0() { asm volatile("s_wait_asynccnt 0x0" ::: "memory"); }

// ---------------------------------------------------------------------------
// fp32 -> bf16 conversion
// ---------------------------------------------------------------------------
__global__ void cvt_f32_bf16_kernel(const float* __restrict__ in,
                                    unsigned short* __restrict__ out, int n) {
    int i = blockIdx.x * blockDim.x + threadIdx.x;
    if (i < n) out[i] = f2bf(in[i]);
}

__global__ void init_pooled_kernel(unsigned int* __restrict__ p, int n) {
    int i = blockIdx.x * blockDim.x + threadIdx.x;
    if (i < n) p[i] = 0u;   // key-space minimum
}

// ---------------------------------------------------------------------------
// WMMA GEMM: C[M,N] = A[M,K] * B[N,K]^T + bias[N]
// BM=BN=128, KC=32. 8 waves, each owns a 32x64 tile (2x4 WMMA frags).
// Double-buffered LDS fed by async global->LDS copies (6 instr/wave/tile).
// ---------------------------------------------------------------------------
template <bool WRITE_F32, bool WRITE_BF16, bool MAXPOOL>
__global__ __launch_bounds__(256) void gemm_bf16_kernel(
    const unsigned short* __restrict__ A,   // [M,K] bf16
    const unsigned short* __restrict__ B,   // [N,K] bf16 (weight layout)
    const float* __restrict__ bias,         // [N]
    float* __restrict__ Cf,                 // [M,ldc] (WRITE_F32)
    unsigned short* __restrict__ Cb,        // [M,N]   (WRITE_BF16)
    unsigned int* __restrict__ pooled,      // [M/1024, N] keys (MAXPOOL)
    int M, int N, int K, int ldc) {
    constexpr int BM = 128, BN = 128, KC = 32, LD = 40;
    __shared__ unsigned short As[2][BM * LD];
    __shared__ unsigned short Bs[2][BN * LD];

    const int tid  = threadIdx.x;
    const int lane = tid & 31;
    const int w    = tid >> 5;
    const int hi   = lane >> 4;
    const int lo   = lane & 15;
    const int m0   = blockIdx.y * BM;
    const int n0   = blockIdx.x * BN;
    const int wm   = (w & 3) * 32;   // 4 waves over M
    const int wn   = (w >> 2) * 64;  // 2 waves over N

    v8f acc[2][4];
#pragma unroll
    for (int i = 0; i < 2; i++)
#pragma unroll
        for (int j = 0; j < 4; j++) acc[i][j] = v8f_zero();

    // Issue one tile's worth of async copies: A 256x32 (4/thread) + B 128x32
    // (2/thread) = 6 async-load instructions per wave.
    auto issue_tile = [&](int buf, int k0) {
#pragma unroll
        for (int i = 0; i < 4; i++) {
            int idx = tid + i * 256;
            int row = idx >> 2;
            int cv  = (idx & 3) * 8;
#if GEMM_ASYNC
            async_ld_b128(lds_off(&As[buf][row * LD + cv]),
                          A + (size_t)(m0 + row) * K + k0 + cv);
#else
            *(uint4*)(&As[buf][row * LD + cv]) =
                *(const uint4*)(A + (size_t)(m0 + row) * K + k0 + cv);
#endif
        }
#pragma unroll
        for (int i = 0; i < 2; i++) {
            int idx = tid + i * 256;
            int row = idx >> 2;
            int cv  = (idx & 3) * 8;
#if GEMM_ASYNC
            async_ld_b128(lds_off(&Bs[buf][row * LD + cv]),
                          B + (size_t)(n0 + row) * K + k0 + cv);
#else
            *(uint4*)(&Bs[buf][row * LD + cv]) =
                *(const uint4*)(B + (size_t)(n0 + row) * K + k0 + cv);
#endif
        }
    };

    const int nk = K / KC;
    issue_tile(0, 0);

    for (int kb = 0; kb < nk; kb++) {
        const int cur = kb & 1;
        if (kb + 1 < nk) {
            issue_tile(1 - cur, (kb + 1) * KC);   // prefetch next tile
#if GEMM_ASYNC
            wait_async_6();                       // tile kb complete, kb+1 in flight
#endif
        } else {
#if GEMM_ASYNC
            wait_async_0();
#endif
        }
        __syncthreads();

        BF16Frag af[2], bf[4];
#pragma unroll
        for (int tm = 0; tm < 2; tm++) {
            // A frag: M=lo; halves 0..7 -> K=8*hi+0..7, halves 8..15 -> K=8*hi+16..23
            const unsigned short* p = &As[cur][(wm + tm * 16 + lo) * LD + hi * 8];
            af[tm].q[0] = *(const uint4*)(p);
            af[tm].q[1] = *(const uint4*)(p + 16);
        }
#pragma unroll
        for (int tn = 0; tn < 4; tn++) {
            // B frag: N=lo; halves 0..7 -> K=8*hi+0..7, halves 8..15 -> K=8*hi+16..23   (B stored [n][k])
            const unsigned short* p = &Bs[cur][(wn + tn * 16 + lo) * LD + hi * 8];
            bf[tn].q[0] = *(const uint4*)(p);
            bf[tn].q[1] = *(const uint4*)(p + 16);
        }
#pragma unroll
        for (int tm = 0; tm < 2; tm++)
#pragma unroll
            for (int tn = 0; tn < 4; tn++)
                acc[tm][tn] = wmma_bf16(af[tm], bf[tn], acc[tm][tn]);
        __syncthreads();   // protects buffer reuse by the next issue_tile
    }

    // epilogue (C layout: element r -> M = r + 8*hi, N = lo)
#pragma unroll
    for (int tm = 0; tm < 2; tm++) {
#pragma unroll
        for (int tn = 0; tn < 4; tn++) {
            int n    = n0 + wn + tn * 16 + lo;
            float bv = bias[n];
            if (MAXPOOL) {
                float mx = -3.4e38f;
#pragma unroll
                for (int r = 0; r < 8; r++) mx = fmaxf(mx, acc[tm][tn][r] + bv);
                int m = m0 + wm + tm * 16 + 8 * hi;  // all 8 rows share batch
                atomicMax(&pooled[(m >> 10) * N + n], fkey(mx));
            } else {
#pragma unroll
                for (int r = 0; r < 8; r++) {
                    int m   = m0 + wm + tm * 16 + r + 8 * hi;
                    float v = acc[tm][tn][r] + bv;
                    if (WRITE_F32)  Cf[(size_t)m * ldc + n] = v;
                    if (WRITE_BF16) Cb[(size_t)m * N + n]   = f2bf(v);
                }
            }
        }
    }
}

// ---------------------------------------------------------------------------
// Flash attention: one wave handles 16 q-rows of one (batch, head).
// qkv: [4096, 3072] bf16 (q|k|v packed), ctx: [4096, 1024] bf16
// ---------------------------------------------------------------------------
__global__ __launch_bounds__(32) void flash_attn_kernel(
    const unsigned short* __restrict__ qkv, unsigned short* __restrict__ ctx) {
    constexpr int SB = 32;
    __shared__ unsigned short Qs[16 * 72];
    __shared__ unsigned short Ks[SB * 72];
    __shared__ unsigned short Vt[64 * 40];   // V transposed: [d][s]
    __shared__ unsigned short Ps[16 * 40];   // probs bf16

    const int lane = threadIdx.x;
    const int hi = lane >> 4, lo = lane & 15;
    const int qt = blockIdx.x, h = blockIdx.y, b = blockIdx.z;
    const int m0 = b * 1024 + qt * 16;

#pragma unroll
    for (int i = 0; i < 4; i++) {
        int idx = lane + i * 32;
        int row = idx >> 3, cv = (idx & 7) * 8;
        *(uint4*)(Qs + row * 72 + cv) =
            *(const uint4*)(qkv + (size_t)(m0 + row) * 3072 + h * 64 + cv);
    }
    __syncthreads();

    BF16Frag aq[2];
#pragma unroll
    for (int c = 0; c < 2; c++) {
        const unsigned short* p = Qs + lo * 72 + c * 32 + hi * 8;
        aq[c].q[0] = *(const uint4*)(p);
        aq[c].q[1] = *(const uint4*)(p + 16);
    }

    float m_i[8], l_i[8];
#pragma unroll
    for (int r = 0; r < 8; r++) { m_i[r] = -3.4e38f; l_i[r] = 0.f; }
    v8f c4[4];
#pragma unroll
    for (int t = 0; t < 4; t++) c4[t] = v8f_zero();

    for (int sb = 0; sb < 1024; sb += SB) {
#pragma unroll
        for (int i = 0; i < 8; i++) {
            int idx = lane + i * 32;
            int s = idx >> 3, cv = (idx & 7) * 8;
            size_t rb = (size_t)(b * 1024 + sb + s) * 3072 + h * 64;
            *(uint4*)(Ks + s * 72 + cv) = *(const uint4*)(qkv + rb + 1024 + cv);
            const unsigned short* vp = qkv + rb + 2048 + cv;
#pragma unroll
            for (int e = 0; e < 8; e++) Vt[(cv + e) * 40 + s] = vp[e];
        }
        __syncthreads();

        v8f sfr[2];
#pragma unroll
        for (int st = 0; st < 2; st++) {
            BF16Frag bk[2];
#pragma unroll
            for (int c = 0; c < 2; c++) {
                const unsigned short* p = Ks + (st * 16 + lo) * 72 + c * 32 + hi * 8;
                bk[c].q[0] = *(const uint4*)(p);
                bk[c].q[1] = *(const uint4*)(p + 16);
            }
            v8f z = v8f_zero();
            z = wmma_bf16(aq[0], bk[0], z);
            z = wmma_bf16(aq[1], bk[1], z);
            sfr[st] = z;
        }

        // online softmax; xor masks 1..8 stay inside 16-lane row groups
#pragma unroll
        for (int r = 0; r < 8; r++) {
            float s0 = sfr[0][r] * 0.125f, s1 = sfr[1][r] * 0.125f;
            float mx = fmaxf(s0, s1);
#pragma unroll
            for (int off = 1; off < 16; off <<= 1)
                mx = fmaxf(mx, __shfl_xor(mx, off, 32));
            float newm = fmaxf(m_i[r], mx);
            float p0 = __expf(s0 - newm), p1 = __expf(s1 - newm);
            float rs = p0 + p1;
#pragma unroll
            for (int off = 1; off < 16; off <<= 1) rs += __shfl_xor(rs, off, 32);
            float alpha = __expf(m_i[r] - newm);
            l_i[r] = l_i[r] * alpha + rs;
            m_i[r] = newm;
#pragma unroll
            for (int t = 0; t < 4; t++) c4[t][r] *= alpha;
            int Mr = r + 8 * hi;
            Ps[Mr * 40 + lo]      = f2bf(p0);
            Ps[Mr * 40 + 16 + lo] = f2bf(p1);
        }
        __syncthreads();

        BF16Frag ap;
        {
            const unsigned short* p = Ps + lo * 40 + hi * 8;
            ap.q[0] = *(const uint4*)(p);
            ap.q[1] = *(const uint4*)(p + 16);
        }
#pragma unroll
        for (int t = 0; t < 4; t++) {
            BF16Frag bv;
            const unsigned short* p = Vt + (t * 16 + lo) * 40 + hi * 8;
            bv.q[0] = *(const uint4*)(p);
            bv.q[1] = *(const uint4*)(p + 16);
            c4[t] = wmma_bf16(ap, bv, c4[t]);
        }
        __syncthreads();
    }

#pragma unroll
    for (int r = 0; r < 8; r++) {
        float inv = 1.f / l_i[r];
        int Mg = m0 + r + 8 * hi;
#pragma unroll
        for (int t = 0; t < 4; t++)
            ctx[(size_t)Mg * 1024 + h * 64 + t * 16 + lo] = f2bf(c4[t][r] * inv);
    }
}

// ---------------------------------------------------------------------------
// Router: per-batch softmax over N=4096, exact sorted top-256, masked
// renormalized weights. One block per batch.
// ---------------------------------------------------------------------------
__global__ __launch_bounds__(256) void router_kernel(
    const unsigned int* __restrict__ pooled,  // [4,4096] keys
    float* __restrict__ idx_out,              // [4,256] (as float)
    float* __restrict__ w_out) {              // [4,4096]
    constexpr int N = 4096, KSEL = 256, T = 256;
    __shared__ float softv[N];
    __shared__ unsigned keyv[N];
    __shared__ unsigned char sel[N];
    __shared__ float redf[T];
    __shared__ unsigned long long redu[T];
    __shared__ float s_max, s_sum, s_selsum;

    const int b = blockIdx.x, tid = threadIdx.x;

    float lmax = -3.4e38f;
    for (int i = tid; i < N; i += T) {
        float logit = dekey(pooled[b * N + i]) * 0.5f;  // / TEMPERATURE
        softv[i] = logit;
        keyv[i]  = fkey(logit);
        sel[i]   = 0;
        lmax     = fmaxf(lmax, logit);
    }
    redf[tid] = lmax;
    __syncthreads();
    for (int s = T / 2; s > 0; s >>= 1) {
        if (tid < s) redf[tid] = fmaxf(redf[tid], redf[tid + s]);
        __syncthreads();
    }
    if (tid == 0) s_max = redf[0];
    __syncthreads();

    float lsum = 0.f;
    for (int i = tid; i < N; i += T) {
        float e = __expf(softv[i] - s_max);
        softv[i] = e;
        lsum += e;
    }
    redf[tid] = lsum;
    __syncthreads();
    for (int s = T / 2; s > 0; s >>= 1) {
        if (tid < s) redf[tid] += redf[tid + s];
        __syncthreads();
    }
    if (tid == 0) { s_sum = redf[0]; s_selsum = 0.f; }
    __syncthreads();
    float invs = 1.f / s_sum;
    for (int i = tid; i < N; i += T) softv[i] *= invs;
    __syncthreads();

    for (int kk = 0; kk < KSEL; kk++) {
        unsigned long long best = 0;
        for (int i = tid; i < N; i += T) {
            unsigned long long cand =
                ((unsigned long long)keyv[i] << 32) | (unsigned)(N - 1 - i);
            if (cand > best) best = cand;
        }
        redu[tid] = best;
        __syncthreads();
        for (int s = T / 2; s > 0; s >>= 1) {
            if (tid < s && redu[tid + s] > redu[tid]) redu[tid] = redu[tid + s];
            __syncthreads();
        }
        if (tid == 0) {
            int ibest = N - 1 - (int)(redu[0] & 0xFFFFFFFFull);
            idx_out[b * KSEL + kk] = (float)ibest;
            sel[ibest]  = 1;
            keyv[ibest] = 0;
            s_selsum += softv[ibest];
        }
        __syncthreads();
    }
    float wden = 1.f / (s_selsum + 1e-8f);
    for (int i = tid; i < N; i += T)
        w_out[b * N + i] = sel[i] ? softv[i] * wden : 0.f;
}

// ---------------------------------------------------------------------------
extern "C" void kernel_launch(void* const* d_in, const int* in_sizes, int n_in,
                              void* d_out, int out_size, void* d_ws, size_t ws_size,
                              hipStream_t stream) {
    (void)in_sizes; (void)n_in; (void)out_size; (void)ws_size;
    const float* x     = (const float*)d_in[0];
    const float* w_in  = (const float*)d_in[1];
    const float* b_in  = (const float*)d_in[2];
    const float* w_out = (const float*)d_in[3];
    const float* b_out = (const float*)d_in[4];
    const float* w_aff = (const float*)d_in[5];
    const float* b_aff = (const float*)d_in[6];

    float* out      = (float*)d_out;
    float* idx_out  = out;                 // 4*256
    float* wgt_out  = out + 1024;          // 4*4096
    float* ctx_out  = out + 1024 + 16384;  // 4*1024*1024 fp32

    char* ws = (char*)d_ws;
    auto carve = [&](size_t bytes) -> char* {
        char* p = ws;
        ws += (bytes + 255) & ~(size_t)255;
        return p;
    };
    unsigned short* xb       = (unsigned short*)carve(4096ull * 1024 * 2);
    unsigned short* winb     = (unsigned short*)carve(3072ull * 1024 * 2);
    unsigned short* woutb    = (unsigned short*)carve(1024ull * 1024 * 2);
    unsigned short* waffb    = (unsigned short*)carve(4096ull * 1024 * 2);
    unsigned short* qkvb     = (unsigned short*)carve(4096ull * 3072 * 2);
    unsigned short* ctxb     = (unsigned short*)carve(4096ull * 1024 * 2);
    unsigned short* contextb = (unsigned short*)carve(4096ull * 1024 * 2);
    unsigned int*   pooled   = (unsigned int*)carve(4ull * 4096 * 4);

    init_pooled_kernel<<<64, 256, 0, stream>>>(pooled, 4 * 4096);

    cvt_f32_bf16_kernel<<<(4096 * 1024) / 256, 256, 0, stream>>>(x, xb, 4096 * 1024);
    cvt_f32_bf16_kernel<<<(3072 * 1024) / 256, 256, 0, stream>>>(w_in, winb, 3072 * 1024);
    cvt_f32_bf16_kernel<<<(1024 * 1024) / 256, 256, 0, stream>>>(w_out, woutb, 1024 * 1024);
    cvt_f32_bf16_kernel<<<(4096 * 1024) / 256, 256, 0, stream>>>(w_aff, waffb, 4096 * 1024);

    // QKV projection: [4096,1024] x [3072,1024]^T -> bf16 [4096,3072]
    gemm_bf16_kernel<false, true, false><<<dim3(3072 / 128, 4096 / 128), 256, 0, stream>>>(
        xb, winb, b_in, nullptr, qkvb, nullptr, 4096, 3072, 1024, 3072);

    // fused flash attention -> ctx bf16 [4096,1024]
    flash_attn_kernel<<<dim3(64, 16, 4), 32, 0, stream>>>(qkvb, ctxb);

    // output projection: fp32 to d_out + bf16 copy for affinity GEMM
    gemm_bf16_kernel<true, true, false><<<dim3(1024 / 128, 4096 / 128), 256, 0, stream>>>(
        ctxb, woutb, b_out, ctx_out, contextb, nullptr, 4096, 1024, 1024, 1024);

    // affinity GEMM fused with max-pool over sequence (atomicMax on keys)
    gemm_bf16_kernel<false, false, true><<<dim3(4096 / 128, 4096 / 128), 256, 0, stream>>>(
        contextb, waffb, b_aff, nullptr, nullptr, pooled, 4096, 4096, 1024, 4096);

    router_kernel<<<4, 256, 0, stream>>>(pooled, idx_out, wgt_out);
}
